// DGI_13297218748904
// MI455X (gfx1250) — matrix operand order repaired and checked by reading, not compiled, not run on hardware
//
#include <hip/hip_runtime.h>

// ---------------------------------------------------------------------------
// Types for CDNA5 WMMA (gfx1250, wave32)
// ---------------------------------------------------------------------------
typedef __attribute__((ext_vector_type(16))) __bf16 v16bf;
typedef __attribute__((ext_vector_type(8)))  float  v8f;
typedef __attribute__((ext_vector_type(8)))  unsigned short ushort8x;
typedef __attribute__((ext_vector_type(4)))  unsigned short ushort4x;
typedef __attribute__((ext_vector_type(4)))  float  float4x;

union Frag16 {
    v16bf    v;
    ushort8x h[2];
    unsigned short u[16];
};

__device__ __forceinline__ unsigned short f2bf(float x) {
    unsigned int u = __float_as_uint(x);
    u += 0x7FFFu + ((u >> 16) & 1u);          // round to nearest even
    return (unsigned short)(u >> 16);
}

// ---------------------------------------------------------------------------
// Small utility kernels
// ---------------------------------------------------------------------------
__global__ void zero_i32_kernel(int* p, long n) {
    long i = (long)blockIdx.x * blockDim.x + threadIdx.x;
    if (i < n) p[i] = 0;
}

// in-degree histogram over dst
__global__ void degree_kernel(const int* ei, int E, int* counts) {
    long e = (long)blockIdx.x * blockDim.x + threadIdx.x;
    if (e < E) atomicAdd(&counts[ei[(long)E + e]], 1);
}

__global__ void dinv_kernel(const int* counts, float* dinv, int n) {
    long i = (long)blockIdx.x * blockDim.x + threadIdx.x;
    if (i < n) dinv[i] = rsqrtf(1.0f + (float)counts[i]);
}

// single-block exclusive scan: counts[0..n) -> row_ptr[0..n], zero cursor
__global__ void scan_kernel(const int* counts, int* row_ptr, int* cursor, int n) {
    __shared__ int buf[1024];
    __shared__ int carry;
    if (threadIdx.x == 0) carry = 0;
    __syncthreads();
    for (int base = 0; base < n; base += 1024) {
        int i = base + (int)threadIdx.x;
        int v = (i < n) ? counts[i] : 0;
        buf[threadIdx.x] = v;
        __syncthreads();
        for (int off = 1; off < 1024; off <<= 1) {
            int t = (threadIdx.x >= (unsigned)off) ? buf[threadIdx.x - off] : 0;
            __syncthreads();
            buf[threadIdx.x] += t;
            __syncthreads();
        }
        int incl = buf[threadIdx.x];
        if (i < n) {
            row_ptr[i] = carry + incl - v;
            cursor[i]  = 0;
        }
        __syncthreads();
        if (threadIdx.x == 1023) carry += buf[1023];
        __syncthreads();
    }
    if (threadIdx.x == 0) row_ptr[n] = carry;
}

// scatter edges into CSR, precompute per-edge norm dinv[src]*dinv[dst]
__global__ void scatter_kernel(const int* ei, int E, const int* row_ptr,
                               int* cursor, const float* dinv,
                               int* col, float* wedge) {
    long e = (long)blockIdx.x * blockDim.x + threadIdx.x;
    if (e < E) {
        int s = ei[e];
        int d = ei[(long)E + e];
        int p = row_ptr[d] + atomicAdd(&cursor[d], 1);
        col[p]   = s;
        wedge[p] = dinv[s] * dinv[d];
    }
}

// W [K][NC] fp32 -> WT [NC][K] bf16
__global__ void transpose_w_kernel(const float* W, unsigned short* WT, int K, int NC) {
    long i = (long)blockIdx.x * blockDim.x + threadIdx.x;
    if (i < (long)K * NC) {
        int k = (int)(i / NC);
        int n = (int)(i % NC);
        WT[(long)n * K + k] = f2bf(W[i]);
    }
}

// act_bf[n][f..f+3] = bf16( x[ perm? perm[n] : n ][f..f+3] )   (b128 read, 8B write)
__global__ void gather_x_kernel(const float* __restrict__ x, const int* __restrict__ perm,
                                unsigned short* __restrict__ act, long n, int F) {
    const int F4 = F >> 2;
    long i = (long)blockIdx.x * blockDim.x + threadIdx.x;
    if (i < n * F4) {
        long row   = i / F4;
        int  local = (int)(i % F4);
        long src   = perm ? (long)perm[row] : row;
        float4x v = *(const float4x*)(x + src * F + local * 4);
        ushort4x o;
        o[0] = f2bf(v[0]); o[1] = f2bf(v[1]); o[2] = f2bf(v[2]); o[3] = f2bf(v[3]);
        *(ushort4x*)(act + row * F + local * 4) = o;
    }
}

// ---------------------------------------------------------------------------
// WMMA bf16 GEMM:  C[M][NC] (fp32) = A[M][KD] (bf16) * WT[NC][KD]^T (bf16)
// 256 threads = 8 waves; each wave owns a 16-row strip, sweeps NC in 16-tiles.
// Fragment layouts per CDNA5 ISA 7.12.2 (16-bit A 16x32 / B 32x16).
// ---------------------------------------------------------------------------
template <int KD>
__global__ __launch_bounds__(256)
void gemm_bf16_kernel(const unsigned short* __restrict__ A,
                      const unsigned short* __restrict__ WT,
                      float* __restrict__ C, long M, int NC) {
    const int wave = threadIdx.x >> 5;
    const int lane = threadIdx.x & 31;
    const long rowBase = ((long)blockIdx.x * 8 + wave) * 16;
    if (rowBase >= M) return;

    const int  half = lane >> 4;      // K half-group (A) / K group (B)
    const long m    = rowBase + (lane & 15);
    const int  ncol = lane & 15;
    constexpr int NK = KD / 32;

    // Hoist all A fragments for this 16-row strip (K = KD)
    Frag16 afrag[NK];
#pragma unroll
    for (int kk = 0; kk < NK; ++kk) {
        const unsigned short* p = A + m * KD + kk * 32 + half * 8;
        afrag[kk].h[0] = *(const ushort8x*)(p);        // K = k0 + half*8 + 0..7
        afrag[kk].h[1] = *(const ushort8x*)(p + 16);   // K = k0 + 16 + half*8 + 0..7
    }

    for (int nb = 0; nb < NC; nb += 16) {
        v8f acc = {};
        const unsigned short* bbase = WT + (long)(nb + ncol) * KD + half * 8;
#pragma unroll
        for (int kk = 0; kk < NK; ++kk) {
            Frag16 bf_;
            bf_.h[0] = *(const ushort8x*)(bbase + kk * 32);       // K = k0 + half*8 + 0..7
            bf_.h[1] = *(const ushort8x*)(bbase + kk * 32 + 16);  // K = k0 + 16 + half*8 + 0..7
            acc = __builtin_amdgcn_wmma_f32_16x16x32_bf16(
                false, afrag[kk].v, false, bf_.v, (short)0, acc, false, false);
        }
        float* cptr = C + rowBase * NC;
#pragma unroll
        for (int r = 0; r < 8; ++r) {
            int ml = r + 8 * half;                      // C/D layout: VGPR r -> M=r (+8 hi)
            cptr[(long)ml * NC + nb + ncol] = acc[r];
        }
    }
}

// ---------------------------------------------------------------------------
// CSR aggregation, float4-vectorized:
//   out = sum_e w[e]*hw[col[e]] + dinv^2*hw[node] + bias
// 256 threads/block; F/4 lanes per node; 256/(F/4) nodes per block.
// b128 gathers keep VMEM issue rate 4x lower; hw rows are L2-resident (192MB).
// ---------------------------------------------------------------------------
__global__ __launch_bounds__(256)
void agg_kernel(const float* __restrict__ hw,
                const int* __restrict__ row_ptr,
                const int* __restrict__ col,
                const float* __restrict__ wedge,
                const float* __restrict__ dinv,
                const float* __restrict__ bias,
                unsigned short* out_bf, float* out_f,
                long N, int F, int relu) {
    const int F4    = F >> 2;                 // lanes per node
    const int npb   = 256 / F4;               // nodes per block
    const int local = (int)threadIdx.x % F4;
    const int slot  = (int)threadIdx.x / F4;
    const long node = (long)blockIdx.x * npb + slot;
    if (node >= N) return;

    const float4x* hw4 = (const float4x*)hw;
    const long rstride = F4;                  // float4 elements per row

    float ax = 0.f, ay = 0.f, az = 0.f, aw = 0.f;
    const int beg = row_ptr[node];
    const int end = row_ptr[node + 1];
    for (int e = beg; e < end; ++e) {
        const float  wv = wedge[e];
        const float4x v = hw4[(long)col[e] * rstride + local];
        ax += wv * v[0]; ay += wv * v[1]; az += wv * v[2]; aw += wv * v[3];
    }
    const float dn = dinv[node];
    const float sn = dn * dn;
    const float4x hself = hw4[node * rstride + local];
    const float4x bvec  = *(const float4x*)(bias + local * 4);
    float vx = ax + sn * hself[0] + bvec[0];
    float vy = ay + sn * hself[1] + bvec[1];
    float vz = az + sn * hself[2] + bvec[2];
    float vw = aw + sn * hself[3] + bvec[3];
    if (relu) {
        vx = fmaxf(vx, 0.f); vy = fmaxf(vy, 0.f);
        vz = fmaxf(vz, 0.f); vw = fmaxf(vw, 0.f);
    }
    if (out_bf) {
        ushort4x o;
        o[0] = f2bf(vx); o[1] = f2bf(vy); o[2] = f2bf(vz); o[3] = f2bf(vw);
        *(ushort4x*)(out_bf + node * F + local * 4) = o;
    }
    if (out_f) {
        float4x o; o[0] = vx; o[1] = vy; o[2] = vz; o[3] = vw;
        *(float4x*)(out_f + node * F + local * 4) = o;
    }
}

// ---------------------------------------------------------------------------
// Summary vector: deterministic two-pass column mean + sigmoid
// ---------------------------------------------------------------------------
__global__ void colsum_partial_kernel(const float* __restrict__ H,
                                      float* __restrict__ partials, long n, int F) {
    const int f = threadIdx.x;
    const long rows_per = (n + gridDim.x - 1) / gridDim.x;
    const long r0 = (long)blockIdx.x * rows_per;
    const long r1 = (r0 + rows_per < n) ? (r0 + rows_per) : n;
    float s = 0.0f;
    for (long r = r0; r < r1; ++r) s += H[r * F + f];
    partials[(long)blockIdx.x * F + f] = s;
}

__global__ void finalize_s_kernel(const float* __restrict__ partials, int nblocks,
                                  float* __restrict__ s_out, long n, int F) {
    const int f = threadIdx.x;
    float s = 0.0f;
    for (int b = 0; b < nblocks; ++b) s += partials[(long)b * F + f];
    s /= (float)n;
    s_out[f] = 1.0f / (1.0f + expf(-s));
}

// Wbs[d] = sum_e Wb[d][e] * s[e]
__global__ void wbs_kernel(const float* __restrict__ Wb, const float* __restrict__ s,
                           float* __restrict__ wbs, int F) {
    const int d = threadIdx.x;
    float acc = 0.0f;
    for (int e = 0; e < F; ++e) acc += Wb[(long)d * F + e] * s[e];
    wbs[d] = acc;
}

// score[n] = dot(H[n,:], wbs) + bb   (one wave32 per node)
__global__ void score_kernel(const float* __restrict__ H, const float* __restrict__ wbs,
                             const float* __restrict__ bb, float* __restrict__ out,
                             long n, int F) {
    const int wave = threadIdx.x >> 5;
    const int lane = threadIdx.x & 31;
    const long node = (long)blockIdx.x * 8 + wave;
    if (node >= n) return;
    const float* h = H + node * F;
    float acc = 0.0f;
    for (int f = lane; f < F; f += 32) acc += h[f] * wbs[f];
#pragma unroll
    for (int off = 16; off > 0; off >>= 1) acc += __shfl_down(acc, off, 32);
    if (lane == 0) out[node] = acc + bb[0];
}

// ---------------------------------------------------------------------------
// Host-side orchestration
// ---------------------------------------------------------------------------
extern "C" void kernel_launch(void* const* d_in, const int* in_sizes, int n_in,
                              void* d_out, int out_size, void* d_ws, size_t ws_size,
                              hipStream_t stream) {
    (void)n_in; (void)out_size; (void)ws_size;
    const float* x    = (const float*)d_in[0];
    const int*   ei   = (const int*)d_in[1];
    const int*   perm = (const int*)d_in[2];
    const float* W1   = (const float*)d_in[3];
    const float* b1   = (const float*)d_in[4];
    const float* W2   = (const float*)d_in[5];
    const float* b2   = (const float*)d_in[6];
    const float* Wb   = (const float*)d_in[7];
    const float* bb   = (const float*)d_in[8];
    float* out = (float*)d_out;

    const long N   = in_sizes[2];              // perm length
    const int  E   = in_sizes[1] / 2;
    const int  IN  = (int)(in_sizes[0] / N);   // 256
    const int  HID = in_sizes[4];              // 256
    const int  OUT = in_sizes[6];              // 128

    // ---- workspace carve-out (256B aligned) ----
    char* w = (char*)d_ws;
    size_t off = 0;
    auto take = [&](size_t bytes) {
        void* p = w + off;
        off = (off + bytes + 255) & ~(size_t)255;
        return p;
    };
    unsigned short* act_bf  = (unsigned short*)take((size_t)N * IN * 2);  // later aliased as hw2
    float*          hw1     = (float*)take((size_t)N * HID * 4);
    unsigned short* act2_bf = (unsigned short*)take((size_t)N * HID * 2);
    float*          H       = (float*)take((size_t)N * OUT * 4);
    unsigned short* W1T     = (unsigned short*)take((size_t)IN * HID * 2);
    unsigned short* W2T     = (unsigned short*)take((size_t)HID * OUT * 2);
    int*   counts  = (int*)take((size_t)N * 4);
    int*   cursor  = (int*)take((size_t)N * 4);
    int*   row_ptr = (int*)take(((size_t)N + 1) * 4);
    float* dinv    = (float*)take((size_t)N * 4);
    int*   col     = (int*)take((size_t)E * 4);
    float* wedge   = (float*)take((size_t)E * 4);
    float* partials= (float*)take((size_t)256 * OUT * 4);
    float* svec    = (float*)take((size_t)OUT * 4);
    float* wbs     = (float*)take((size_t)OUT * 4);
    float* hw2     = (float*)act_bf;  // alias: act_bf dead once hw2 is produced

    const int T = 256;
    const unsigned gN    = (unsigned)((N + T - 1) / T);
    const unsigned gE    = (unsigned)(((long)E + T - 1) / T);
    const unsigned gNF4  = (unsigned)(((long)N * (IN / 4) + T - 1) / T);
    const unsigned gW1   = (unsigned)(((long)IN * HID + T - 1) / T);
    const unsigned gW2   = (unsigned)(((long)HID * OUT + T - 1) / T);
    const unsigned gGem  = (unsigned)((N + 127) / 128);
    const unsigned gScr  = (unsigned)((N + 7) / 8);
    const unsigned gAgg1 = (unsigned)((N + (256 / (HID / 4)) - 1) / (256 / (HID / 4)));
    const unsigned gAgg2 = (unsigned)((N + (256 / (OUT / 4)) - 1) / (256 / (OUT / 4)));

    // ---- CSR build (per call; no persistent state) ----
    zero_i32_kernel<<<gN, T, 0, stream>>>(counts, N);
    degree_kernel<<<gE, T, 0, stream>>>(ei, E, counts);
    dinv_kernel<<<gN, T, 0, stream>>>(counts, dinv, (int)N);
    scan_kernel<<<1, 1024, 0, stream>>>(counts, row_ptr, cursor, (int)N);
    scatter_kernel<<<gE, T, 0, stream>>>(ei, E, row_ptr, cursor, dinv, col, wedge);

    // ---- weight transpose + bf16 convert ----
    transpose_w_kernel<<<gW1, T, 0, stream>>>(W1, W1T, IN, HID);
    transpose_w_kernel<<<gW2, T, 0, stream>>>(W2, W2T, HID, OUT);

    // ================= positive path =================
    gather_x_kernel<<<gNF4, T, 0, stream>>>(x, nullptr, act_bf, N, IN);
    gemm_bf16_kernel<256><<<gGem, 256, 0, stream>>>(act_bf, W1T, hw1, N, HID);
    agg_kernel<<<gAgg1, 256, 0, stream>>>(hw1, row_ptr, col, wedge, dinv, b1,
                                          act2_bf, nullptr, N, HID, 1);
    gemm_bf16_kernel<256><<<gGem, 256, 0, stream>>>(act2_bf, W2T, hw2, N, OUT);
    agg_kernel<<<gAgg2, 256, 0, stream>>>(hw2, row_ptr, col, wedge, dinv, b2,
                                          nullptr, H, N, OUT, 0);

    // summary + bilinear vector
    colsum_partial_kernel<<<256, OUT, 0, stream>>>(H, partials, N, OUT);
    finalize_s_kernel<<<1, OUT, 0, stream>>>(partials, 256, svec, N, OUT);
    wbs_kernel<<<1, OUT, 0, stream>>>(Wb, svec, wbs, OUT);
    score_kernel<<<gScr, 256, 0, stream>>>(H, wbs, bb, out, N, OUT);

    // ================= negative path =================
    gather_x_kernel<<<gNF4, T, 0, stream>>>(x, perm, act_bf, N, IN);
    gemm_bf16_kernel<256><<<gGem, 256, 0, stream>>>(act_bf, W1T, hw1, N, HID);
    agg_kernel<<<gAgg1, 256, 0, stream>>>(hw1, row_ptr, col, wedge, dinv, b1,
                                          act2_bf, nullptr, N, HID, 1);
    gemm_bf16_kernel<256><<<gGem, 256, 0, stream>>>(act2_bf, W2T, hw2, N, OUT);
    agg_kernel<<<gAgg2, 256, 0, stream>>>(hw2, row_ptr, col, wedge, dinv, b2,
                                          nullptr, H, N, OUT, 0);
    score_kernel<<<gScr, 256, 0, stream>>>(H, wbs, bb, out + N, N, OUT);
}
